// GIN_22574348108106
// MI455X (gfx1250) — matrix operand order repaired and checked by reading, not compiled, not run on hardware
//
#include <hip/hip_runtime.h>
#include <hip/hip_bf16.h>

typedef __attribute__((ext_vector_type(16))) __bf16 v16bf;
typedef __attribute__((ext_vector_type(8)))  float  v8f;

// ---------------------------------------------------------------------------
// Generic utility kernels
// ---------------------------------------------------------------------------
__global__ void k_zero(float* __restrict__ p, int n) {
    int i = blockIdx.x * blockDim.x + threadIdx.x;
    int stride = gridDim.x * blockDim.x;
    for (; i < n; i += stride) p[i] = 0.0f;
}

__global__ void k_copy4(float* __restrict__ dst, const float* __restrict__ src, int n4) {
    int i = blockIdx.x * blockDim.x + threadIdx.x;
    if (i < n4) ((float4*)dst)[i] = ((const float4*)src)[i];
}

// ---------------------------------------------------------------------------
// Convert a row-major f32 weight [K,64] into bf16 WMMA B-fragments.
// Fragment (kb, nb) covers K rows [kb*32, kb*32+32), N cols [nb*16, nb*16+16).
// Layout per ISA 7.12.2 (16-bit B 32x16): lane L<16 -> col N=nb*16+L, K=kb*32+(j<8 ? j : j+8)
//                                         lane L>=16 -> col N=nb*16+(L-16), K=kb*32+8+(j<8 ? j : j+8)
// Stored lane-major: dst[((kb*4+nb)*32 + lane)*16 + j]
// ---------------------------------------------------------------------------
__global__ void k_convert_wfrag(const float* __restrict__ W, __bf16* __restrict__ dst, int K) {
    int idx = blockIdx.x * blockDim.x + threadIdx.x;
    int total = K * 64;
    if (idx >= total) return;
    int j    = idx & 15;
    int lane = (idx >> 4) & 31;
    int fr   = idx >> 9;           // kb*4 + nb
    int kb   = fr >> 2;
    int nb   = fr & 3;
    int k = kb * 32 + ((lane >> 4) << 3) + (j < 8 ? j : j + 8);
    int n = nb * 16 + (lane & 15);
    dst[idx] = (__bf16)W[k * 64 + n];
}

// ---------------------------------------------------------------------------
// Fused GEMM: y[N,64] = act[N,K] @ W + bias, plus per-column sum / sum^2
// stats (for BatchNorm).  One wave computes a 16x64 output tile with 4
// v_wmma_f32_16x16x32_bf16 per 32-wide K block.  A fragment built on the fly
// from f32 activations (ISA 16-bit A 16x32 layout).
// ---------------------------------------------------------------------------
__global__ __launch_bounds__(256)
void k_gemm_stats(const float* __restrict__ act, const __bf16* __restrict__ wfrag,
                  const float* __restrict__ bias, float* __restrict__ y,
                  float* __restrict__ stats /* [64]=sum, [64..128)=sumsq */,
                  int K, int ntiles) {
    __shared__ float lsum[64];
    __shared__ float lsq[64];
    if (threadIdx.x < 64) { lsum[threadIdx.x] = 0.0f; lsq[threadIdx.x] = 0.0f; }
    __syncthreads();

    const int wave = threadIdx.x >> 5;
    const int lane = threadIdx.x & 31;
    const int tile = blockIdx.x * 8 + wave;

    float s[4]  = {0.f, 0.f, 0.f, 0.f};
    float s2[4] = {0.f, 0.f, 0.f, 0.f};

    if (tile < ntiles) {
        const int m     = tile * 16 + (lane & 15);       // A row for this lane
        const int kbase = (lane >> 4) << 3;              // 0 or 8
        const int kblocks = K >> 5;

        v8f acc0 = {}, acc1 = {}, acc2 = {}, acc3 = {};
        for (int kb = 0; kb < kblocks; ++kb) {
            const float* ap = act + (size_t)m * K + kb * 32 + kbase;
            v16bf a;
#pragma unroll
            for (int j = 0; j < 8; ++j) a[j] = (__bf16)ap[j];        // K = kbase + 0..7
#pragma unroll
            for (int j = 0; j < 8; ++j) a[8 + j] = (__bf16)ap[16 + j]; // K = kbase + 16..23

            const v16bf* bp = (const v16bf*)wfrag + (size_t)(kb * 4) * 32 + lane;
            v16bf b0 = bp[0];
            v16bf b1 = bp[32];
            v16bf b2 = bp[64];
            v16bf b3 = bp[96];
            acc0 = __builtin_amdgcn_wmma_f32_16x16x32_bf16(false, a, false, b0, (short)0, acc0, false, false);
            acc1 = __builtin_amdgcn_wmma_f32_16x16x32_bf16(false, a, false, b1, (short)0, acc1, false, false);
            acc2 = __builtin_amdgcn_wmma_f32_16x16x32_bf16(false, a, false, b2, (short)0, acc2, false, false);
            acc3 = __builtin_amdgcn_wmma_f32_16x16x32_bf16(false, a, false, b3, (short)0, acc3, false, false);
        }

        // C/D layout: VGPR r -> row (tile*16 + r + (lane>=16 ? 8 : 0)), col = nb*16 + (lane&15)
        const int col    = lane & 15;
        const int rowoff = (lane >> 4) << 3;
        const int row0   = tile * 16 + rowoff;
        v8f accs[4] = {acc0, acc1, acc2, acc3};
#pragma unroll
        for (int nb = 0; nb < 4; ++nb) {
            const int c = nb * 16 + col;
            const float bv = bias[c];
#pragma unroll
            for (int r = 0; r < 8; ++r) {
                float v = accs[nb][r] + bv;
                y[(size_t)(row0 + r) * 64 + c] = v;
                s[nb]  += v;
                s2[nb] += v * v;
            }
        }
    }

    // block-level column reduction, then one global atomic per column
    {
        const int col = lane & 15;
#pragma unroll
        for (int nb = 0; nb < 4; ++nb) {
            const int c = nb * 16 + col;
            atomicAdd(&lsum[c], s[nb]);
            atomicAdd(&lsq[c], s2[nb]);
        }
    }
    __syncthreads();
    if (threadIdx.x < 64) {
        atomicAdd(&stats[threadIdx.x],      lsum[threadIdx.x]);
        atomicAdd(&stats[64 + threadIdx.x], lsq[threadIdx.x]);
    }
}

// ---------------------------------------------------------------------------
// BatchNorm fold: scale = g * rsqrt(var + eps); shift = bt - mean*scale
// ---------------------------------------------------------------------------
__global__ void k_bn_finalize(const float* __restrict__ stats, const float* __restrict__ g,
                              const float* __restrict__ bt, float* __restrict__ ss, float invn) {
    int c = threadIdx.x;
    if (c < 64) {
        float mean = stats[c] * invn;
        float var  = stats[64 + c] * invn - mean * mean;
        float sc   = g[c] * rsqrtf(var + 1e-5f);
        ss[c]      = sc;
        ss[64 + c] = bt[c] - mean * sc;
    }
}

// h = relu(y*scale + shift), vectorized float4 over [N,64]
__global__ void k_bn_relu(const float* __restrict__ y, const float* __restrict__ ss,
                          float* __restrict__ h, int n4) {
    int i = blockIdx.x * blockDim.x + threadIdx.x;
    if (i >= n4) return;
    int c4 = (i & 15) * 4;
    float4 v = ((const float4*)y)[i];
    float4 r;
    r.x = fmaxf(0.f, v.x * ss[c4 + 0] + ss[64 + c4 + 0]);
    r.y = fmaxf(0.f, v.y * ss[c4 + 1] + ss[64 + c4 + 1]);
    r.z = fmaxf(0.f, v.z * ss[c4 + 2] + ss[64 + c4 + 2]);
    r.w = fmaxf(0.f, v.w * ss[c4 + 3] + ss[64 + c4 + 3]);
    ((float4*)h)[i] = r;
}

// ---------------------------------------------------------------------------
// Edge aggregation: agg[dst] += h[src] (agg pre-initialized to h => z = h+agg)
// 16 threads per edge, float4 gather + 4 f32 atomics.
// ---------------------------------------------------------------------------
__global__ __launch_bounds__(256)
void k_edge_agg(float* __restrict__ agg, const float* __restrict__ h,
                const int* __restrict__ src, const int* __restrict__ dst, int E) {
    int t = blockIdx.x * blockDim.x + threadIdx.x;
    int e = t >> 4;
    if (e >= E) return;
    int f = (t & 15) * 4;
    int s = src[e], d = dst[e];
    float4 v = *(const float4*)(h + (size_t)s * 64 + f);
    float* dp = agg + (size_t)d * 64 + f;
    atomicAdd(dp + 0, v.x);
    atomicAdd(dp + 1, v.y);
    atomicAdd(dp + 2, v.z);
    atomicAdd(dp + 3, v.w);
}

// pooled[batch[i]] += h[i,:]; optionally node counts per graph
__global__ __launch_bounds__(256)
void k_pool(const float* __restrict__ h, const int* __restrict__ batch,
            float* __restrict__ pooled, float* __restrict__ counts,
            int N, int do_counts) {
    int t = blockIdx.x * blockDim.x + threadIdx.x;
    int i = t >> 4;
    if (i >= N) return;
    int f = (t & 15) * 4;
    int g = batch[i];
    float4 v = *(const float4*)(h + (size_t)i * 64 + f);
    float* dp = pooled + (size_t)g * 64 + f;
    atomicAdd(dp + 0, v.x);
    atomicAdd(dp + 1, v.y);
    atomicAdd(dp + 2, v.z);
    atomicAdd(dp + 3, v.w);
    if (do_counts && f == 0) atomicAdd(counts + g, 1.0f);
}

// out[g,t] += sum_c pooled[g,c]*linW[c,t] + (use_counts ? counts[g]*linb[t] : linb[t])
__global__ void k_readout(const float* __restrict__ pooled, const float* __restrict__ counts,
                          const float* __restrict__ linW, const float* __restrict__ linb,
                          float* __restrict__ out, int use_counts) {
    int idx = blockIdx.x * blockDim.x + threadIdx.x;
    if (idx >= 512 * 64) return;
    int g = idx >> 6, t = idx & 63;
    const float* pr = pooled + (size_t)g * 64;
    float acc = 0.0f;
#pragma unroll 8
    for (int c = 0; c < 64; ++c) acc += pr[c] * linW[c * 64 + t];
    acc += use_counts ? counts[g] * linb[t] : linb[t];
    out[idx] += acc;
}

// ---------------------------------------------------------------------------
// Host-side orchestration
// ---------------------------------------------------------------------------
extern "C" void kernel_launch(void* const* d_in, const int* in_sizes, int n_in,
                              void* d_out, int out_size, void* d_ws, size_t ws_size,
                              hipStream_t stream) {
    const float* x     = (const float*)d_in[0];
    const int*   ei    = (const int*)  d_in[1];
    const int*   batch = (const int*)  d_in[2];
    const float* fW1  = (const float*)d_in[3];
    const float* fb1  = (const float*)d_in[4];
    const float* fg1  = (const float*)d_in[5];
    const float* fbt1 = (const float*)d_in[6];
    const float* fW2  = (const float*)d_in[7];
    const float* fb2  = (const float*)d_in[8];
    const float* fg2  = (const float*)d_in[9];
    const float* fbt2 = (const float*)d_in[10];
    const float* cW1  = (const float*)d_in[11];
    const float* cb1  = (const float*)d_in[12];
    const float* cg1  = (const float*)d_in[13];
    const float* cbt1 = (const float*)d_in[14];
    const float* cW2  = (const float*)d_in[15];
    const float* cb2  = (const float*)d_in[16];
    const float* cg2  = (const float*)d_in[17];
    const float* cbt2 = (const float*)d_in[18];
    const float* linW = (const float*)d_in[19];
    const float* linb = (const float*)d_in[20];
    float* out = (float*)d_out;

    const int N = in_sizes[0] / 128;   // 100000
    const int E = in_sizes[1] / 2;     // 1600000
    const int G = 512;
    const int NL = 4;
    const float invn = 1.0f / (float)N;

    // ---- workspace layout (floats) ----
    float* ws = (float*)d_ws;
    float* y      = ws;                       // N*64
    float* h      = y   + (size_t)N * 64;     // N*64
    float* agg    = h   + (size_t)N * 64;     // N*64
    float* stats  = agg + (size_t)N * 64;     // 10*128
    float* ss     = stats + 10 * 128;         // 128 (scale, shift)
    float* pooled = ss + 128;                 // 5*G*64
    float* counts = pooled + (size_t)5 * G * 64; // G
    __bf16* wfrag = (__bf16*)(counts + G);
    __bf16* wfW1 = wfrag;                     // 128*64
    __bf16* wfW2 = wfW1 + 128 * 64;           // 64*64
    __bf16* wcW1 = wfW2 + 64 * 64;            // 4*64*64
    __bf16* wcW2 = wcW1 + 4 * 64 * 64;        // 4*64*64

    const int TB = 256;
    const int ntiles  = N / 16;                         // N % 16 == 0
    const int gemmGrid = (ntiles + 7) / 8;
    const int applyN4 = N * 16;
    const int applyGrid = (applyN4 + TB - 1) / TB;
    const int edgeGrid  = ((size_t)E * 16 + TB - 1) / TB;
    const int poolGrid  = ((size_t)N * 16 + TB - 1) / TB;
    const int outGrid   = (G * 64 + TB - 1) / TB;

    // ---- zero accumulators (stats + ss + pooled + counts) and output ----
    int accumN = 10 * 128 + 128 + 5 * G * 64 + G;
    k_zero<<<(accumN + TB - 1) / TB, TB, 0, stream>>>(stats, accumN);
    k_zero<<<(G * 64 + TB - 1) / TB, TB, 0, stream>>>(out, G * 64);

    // ---- convert weights to bf16 WMMA B-fragments ----
    k_convert_wfrag<<<(128 * 64 + TB - 1) / TB, TB, 0, stream>>>(fW1, wfW1, 128);
    k_convert_wfrag<<<(64 * 64 + TB - 1) / TB, TB, 0, stream>>>(fW2, wfW2, 64);
    for (int l = 0; l < NL; ++l) {
        k_convert_wfrag<<<(64 * 64 + TB - 1) / TB, TB, 0, stream>>>(cW1 + (size_t)l * 4096, wcW1 + (size_t)l * 4096, 64);
        k_convert_wfrag<<<(64 * 64 + TB - 1) / TB, TB, 0, stream>>>(cW2 + (size_t)l * 4096, wcW2 + (size_t)l * 4096, 64);
    }

    // ---- first_h MLP ----
    k_gemm_stats<<<gemmGrid, TB, 0, stream>>>(x, wfW1, fb1, y, stats + 0 * 128, 128, ntiles);
    k_bn_finalize<<<1, 64, 0, stream>>>(stats + 0 * 128, fg1, fbt1, ss, invn);
    k_bn_relu<<<applyGrid, TB, 0, stream>>>(y, ss, h, applyN4);

    k_gemm_stats<<<gemmGrid, TB, 0, stream>>>(h, wfW2, fb2, y, stats + 1 * 128, 64, ntiles);
    k_bn_finalize<<<1, 64, 0, stream>>>(stats + 1 * 128, fg2, fbt2, ss, invn);
    k_bn_relu<<<applyGrid, TB, 0, stream>>>(y, ss, h, applyN4);

    // layer-0 readout: segment_sum(h@W0 + b0) == pool(h)@W0 + counts*b0
    k_pool<<<poolGrid, TB, 0, stream>>>(h, batch, pooled, counts, N, 1);
    k_readout<<<outGrid, TB, 0, stream>>>(pooled, counts, linW, linb, out, 1);

    // ---- GIN layers ----
    const int* srcIdx = ei;
    const int* dstIdx = ei + E;
    for (int l = 0; l < NL; ++l) {
        // z = h + sum_{j->i} h_j : init agg = h, then atomic edge adds
        k_copy4<<<applyGrid, TB, 0, stream>>>(agg, h, applyN4);
        k_edge_agg<<<edgeGrid, TB, 0, stream>>>(agg, h, srcIdx, dstIdx, E);

        // MLP layer 1
        k_gemm_stats<<<gemmGrid, TB, 0, stream>>>(agg, wcW1 + (size_t)l * 4096, cb1 + l * 64,
                                                  y, stats + (2 + 2 * l) * 128, 64, ntiles);
        k_bn_finalize<<<1, 64, 0, stream>>>(stats + (2 + 2 * l) * 128, cg1 + l * 64, cbt1 + l * 64, ss, invn);
        k_bn_relu<<<applyGrid, TB, 0, stream>>>(y, ss, agg, applyN4);   // agg now holds intermediate z

        // MLP layer 2
        k_gemm_stats<<<gemmGrid, TB, 0, stream>>>(agg, wcW2 + (size_t)l * 4096, cb2 + l * 64,
                                                  y, stats + (3 + 2 * l) * 128, 64, ntiles);
        k_bn_finalize<<<1, 64, 0, stream>>>(stats + (3 + 2 * l) * 128, cg2 + l * 64, cbt2 + l * 64, ss, invn);
        k_bn_relu<<<applyGrid, TB, 0, stream>>>(y, ss, h, applyN4);     // new h

        // readout
        float* pl = pooled + (size_t)(l + 1) * G * 64;
        k_pool<<<poolGrid, TB, 0, stream>>>(h, batch, pl, counts, N, 0);
        k_readout<<<outGrid, TB, 0, stream>>>(pl, counts,
                                              linW + (size_t)(l + 1) * 4096,
                                              linb + (size_t)(l + 1) * 64, out, 0);
    }
}
